// ChannelAttention_39281770889704
// MI455X (gfx1250) — matrix operand order repaired and checked by reading, not compiled, not run on hardware
//
#include <hip/hip_runtime.h>
#include <hip/hip_bf16.h>
#include <math.h>
#include <stdint.h>

typedef __attribute__((ext_vector_type(16))) __bf16 v16bf;
typedef __attribute__((ext_vector_type(8)))  float  v8f;
typedef __attribute__((ext_vector_type(2)))  float  v2f;
typedef __attribute__((ext_vector_type(4)))  unsigned int v4u;
typedef __attribute__((ext_vector_type(8)))  int    v8i;
typedef __attribute__((ext_vector_type(4)))  int    v4i;

union BF16x16 { unsigned short u[16]; v16bf v; };

static __device__ inline unsigned short f2bf(float f) {
    unsigned int u = __float_as_uint(f);
    u += 0x7fffu + ((u >> 16) & 1u);      // round-to-nearest-even
    return (unsigned short)(u >> 16);
}

// Tensor Data Mover availability (device pass only; host pass sees 0)
#if defined(__has_builtin)
#if __has_builtin(__builtin_amdgcn_tensor_load_to_lds) && __has_builtin(__builtin_amdgcn_s_wait_tensorcnt)
#define USE_TDM 1
#endif
#endif
#ifndef USE_TDM
#define USE_TDM 0
#endif

#if USE_TDM
// ---------------------------------------------------------------------------
// Issue one TDM 2-D tile load: tile_y rows of tile_x fp32 elements, global
// row stride = row_stride_elems, packed contiguously into LDS at lds_byte_off.
// D# layout per CDNA5 ISA 8.3 (group0) / 8.4 (group1); groups 2/3 zero (2-D).
// ---------------------------------------------------------------------------
__device__ __forceinline__ void tdm_load_tile_2d(
    unsigned lds_byte_off, const float* gptr,
    unsigned tile_x, unsigned tile_y, unsigned long long row_stride_elems,
    unsigned tensor_d0, unsigned tensor_d1)
{
    unsigned long long ga = (unsigned long long)(uintptr_t)gptr;
    v4u g0;
    g0[0] = 1u;                                         // count=1, user mode
    g0[1] = lds_byte_off;                               // lds_addr
    g0[2] = (unsigned)ga;                               // global_addr[31:0]
    g0[3] = (unsigned)((ga >> 32) & 0x1FFFFFFull)       // global_addr[56:32]
            | 0x80000000u;                              // type=2 ("image")
    v8i g1;
    g1[0] = 0x00020000;                                 // data_size=4B, no pad
    g1[1] = (int)((tensor_d0 & 0xFFFFu) << 16);         // tensor_dim0[15:0]
    g1[2] = (int)((tensor_d0 >> 16) | ((tensor_d1 & 0xFFFFu) << 16));
    g1[3] = (int)((tensor_d1 >> 16) | (tile_x << 16));  // tile_dim0
    g1[4] = (int)tile_y;                                // tile_dim1 (dim2=0)
    g1[5] = (int)(unsigned)(row_stride_elems & 0xFFFFFFFFull);
    g1[6] = (int)(unsigned)((row_stride_elems >> 32) & 0xFFFFull);
    g1[7] = 0;                                          // dim1 stride unused
    v4i z4; z4[0] = z4[1] = z4[2] = z4[3] = 0;
#if defined(__clang_major__) && (__clang_major__ >= 23)
    v8i z8; for (int i = 0; i < 8; i++) z8[i] = 0;
    __builtin_amdgcn_tensor_load_to_lds(g0, g1, z4, z4, z8, 0);
#else
    __builtin_amdgcn_tensor_load_to_lds(g0, g1, z4, z4, 0);
#endif
}
#endif

// ---------------------------------------------------------------------------
// Pointwise-conv GEMM:  C[b] = A (MxK fp32 weights) * B[b] (KxN fp32 acts)
// bf16 WMMA, fp32 accumulate. Block tile 64(M) x 128(N), BK=32, 8 waves.
// Tiles staged by the Tensor Data Mover (double buffered), converted to bf16
// in LDS, fragments per ISA 7.12.2 layouts.
// ---------------------------------------------------------------------------
#define TM 64
#define TN 128
#define TK 32

__global__ __launch_bounds__(256)
void gemm_bf16_nn(const float* __restrict__ A, const float* __restrict__ Bm,
                  float* __restrict__ Cm, int M, int K, int N,
                  long long strideB, long long strideC)
{
    __shared__ unsigned short As[TM][TK];        // 4 KB   bf16 A tile
    __shared__ unsigned short Bs[TN][TK + 2];    // 8.5 KB bf16 B tile (transposed)
#if USE_TDM
    __shared__ float Ast[2][TM][TK];             // 16 KB  fp32 TDM stage
    __shared__ float Bst[2][TK][TN];             // 32 KB  fp32 TDM stage
#endif

    const int t    = threadIdx.x;
    const int lane = t & 31;
    const int wave = t >> 5;
    const int m0   = blockIdx.y * TM;
    const int n0   = blockIdx.x * TN;
    const float* Bp = Bm + (long long)blockIdx.z * strideB;
    float*       Cp = Cm + (long long)blockIdx.z * strideC;

    const int wm = (wave >> 2) * 32;   // 0 or 32
    const int wn = (wave & 3) * 32;    // 0..96

    v8f acc[2][2];
    #pragma unroll
    for (int i = 0; i < 2; i++)
        #pragma unroll
        for (int j = 0; j < 2; j++)
            #pragma unroll
            for (int e = 0; e < 8; e++) acc[i][j][e] = 0.f;

    const int mrow = lane & 15;
    const int kbA  = (lane < 16) ? 0 : 8;
    const int kbB  = (lane < 16) ? 0 : 8;

#if USE_TDM
    const int nk = K / TK;
    if (wave == 0) {
        tdm_load_tile_2d((unsigned)(uintptr_t)(void*)&Ast[0][0][0],
                         A + (long long)m0 * K, TK, TM,
                         (unsigned long long)K, (unsigned)K, (unsigned)M);
        tdm_load_tile_2d((unsigned)(uintptr_t)(void*)&Bst[0][0][0],
                         Bp + n0, TN, TK,
                         (unsigned long long)N, (unsigned)N, (unsigned)K);
    }
    for (int kidx = 0; kidx < nk; kidx++) {
        const int p = kidx & 1;
        if (wave == 0) __builtin_amdgcn_s_wait_tensorcnt(0);
        __syncthreads();                          // stage[p] ready for all waves
        if (wave == 0 && kidx + 1 < nk) {         // overlap next DMA with compute
            const int k1 = (kidx + 1) * TK;
            tdm_load_tile_2d((unsigned)(uintptr_t)(void*)&Ast[p ^ 1][0][0],
                             A + (long long)m0 * K + k1, TK, TM,
                             (unsigned long long)K, (unsigned)K, (unsigned)M);
            tdm_load_tile_2d((unsigned)(uintptr_t)(void*)&Bst[p ^ 1][0][0],
                             Bp + (long long)k1 * N + n0, TN, TK,
                             (unsigned long long)N, (unsigned)N, (unsigned)K);
        }
        // convert fp32 stage -> bf16 operand tiles
        #pragma unroll
        for (int r = 0; r < 8; r++) {
            int idx = t + r * 256;
            int row = idx >> 5, kk = idx & 31;
            As[row][kk] = f2bf(Ast[p][row][kk]);
        }
        #pragma unroll
        for (int r = 0; r < 16; r++) {
            int idx = t + r * 256;
            int kk  = idx >> 7, nn = idx & 127;
            Bs[nn][kk] = f2bf(Bst[p][kk][nn]);
        }
        __syncthreads();
#else
    for (int k0 = 0; k0 < K; k0 += TK) {
        // stage A tile (64x32) as bf16, coalesced over K
        #pragma unroll
        for (int r = 0; r < 8; r++) {
            int idx = t + r * 256;
            int row = idx >> 5, kk = idx & 31;
            As[row][kk] = f2bf(A[(long long)(m0 + row) * K + (k0 + kk)]);
        }
        // stage B tile (32x128) transposed into LDS, coalesced over N
        #pragma unroll
        for (int r = 0; r < 16; r++) {
            int idx = t + r * 256;
            int kk  = idx >> 7;
            int nn  = idx & 127;
            Bs[nn][kk] = f2bf(Bp[(long long)(k0 + kk) * N + (n0 + nn)]);
        }
        __syncthreads();
#endif
        BF16x16 afr[2], bfr[2];
        #pragma unroll
        for (int i = 0; i < 2; i++) {
            int row = wm + i * 16 + mrow;
            #pragma unroll
            for (int e = 0; e < 16; e++)   // 16-bit A 16x32 layout (ISA 7.12.2)
                afr[i].u[e] = As[row][kbA + e + ((e >= 8) ? 8 : 0)];
        }
        #pragma unroll
        for (int j = 0; j < 2; j++) {
            int col = wn + j * 16 + mrow;
            #pragma unroll
            for (int e = 0; e < 16; e++)   // B 32x16: half-wave K split
                bfr[j].u[e] = Bs[col][kbB + e + ((e >= 8) ? 8 : 0)];
        }
        #pragma unroll
        for (int i = 0; i < 2; i++)
            #pragma unroll
            for (int j = 0; j < 2; j++)
                acc[i][j] = __builtin_amdgcn_wmma_f32_16x16x32_bf16(
                    false, afr[i].v, false, bfr[j].v,
                    (short)0, acc[i][j], false, false);
        __syncthreads();
    }

    const int ccol = lane & 15;
    const int radd = (lane < 16) ? 0 : 8;
    #pragma unroll
    for (int i = 0; i < 2; i++)
        #pragma unroll
        for (int j = 0; j < 2; j++)
            #pragma unroll
            for (int v = 0; v < 8; v++) {
                int row = m0 + wm + i * 16 + radd + v;
                int col = n0 + wn + j * 16 + ccol;
                Cp[(long long)row * N + col] = acc[i][j][v];
            }
}

// ---------------------------------------------------------------------------
// Depthwise 3x3, pad 1, optional exact GELU
// ---------------------------------------------------------------------------
__global__ __launch_bounds__(256)
void dw3x3(const float* __restrict__ in, const float* __restrict__ w,
           float* __restrict__ out, int Cc, int H, int W,
           int outBatchCh, int outChOff, int act)
{
    int n = blockIdx.x * 256 + threadIdx.x;
    int c = blockIdx.y, b = blockIdx.z;
    int hh = n / W, ww = n % W;
    const float* ip = in + ((long long)b * Cc + c) * (long long)(H * W);
    const float* wp = w + c * 9;
    float s = 0.f;
    #pragma unroll
    for (int dy = -1; dy <= 1; dy++)
        #pragma unroll
        for (int dx = -1; dx <= 1; dx++) {
            int y = hh + dy, x = ww + dx;
            if (y >= 0 && y < H && x >= 0 && x < W)
                s = fmaf(wp[(dy + 1) * 3 + (dx + 1)], ip[y * W + x], s);
        }
    if (act) s = 0.5f * s * (1.f + erff(s * 0.70710678118654752f));
    out[((long long)b * outBatchCh + outChOff + c) * (long long)(H * W) + n] = s;
}

// ---------------------------------------------------------------------------
// Per-row 1/max(||row||,1e-12) for q (y=0) and k (y=1)
// ---------------------------------------------------------------------------
__global__ __launch_bounds__(256)
void row_invnorm(const float* __restrict__ qkv, float* __restrict__ invq,
                 float* __restrict__ invk, int C, int N)
{
    int x = blockIdx.x;              // b*C + c
    int which = blockIdx.y;
    int b = x / C, c = x % C;
    const float* row = qkv + ((long long)b * 3 * C + which * C + c) * (long long)N;
    float s = 0.f;
    for (int n = threadIdx.x; n < N; n += 256) { float v = row[n]; s = fmaf(v, v, s); }
    __shared__ float sm[256];
    sm[threadIdx.x] = s; __syncthreads();
    for (int off = 128; off > 0; off >>= 1) {
        if (threadIdx.x < off) sm[threadIdx.x] += sm[threadIdx.x + off];
        __syncthreads();
    }
    if (threadIdx.x == 0)
        (which == 0 ? invq : invk)[x] = 1.f / fmaxf(sqrtf(sm[0]), 1e-12f);
}

// ---------------------------------------------------------------------------
// attn logits: S = scale * diag(invq) (q . k^T) diag(invk), fp32 WMMA 16x16x4,
// K=16384 split over 8 waves, LDS tree-reduce epilogue.
// ---------------------------------------------------------------------------
__global__ __launch_bounds__(256)
void attn_qk(const float* __restrict__ qkv, const float* __restrict__ invq,
             const float* __restrict__ invk, float* __restrict__ attn,
             int C, int N, float scale)
{
    const int t = threadIdx.x, lane = t & 31, wave = t >> 5;
    const int bh = blockIdx.z;                 // b*8 + h
    const int b = bh >> 3, h = bh & 7;
    const int c0 = blockIdx.y * 16;
    const int d0 = blockIdx.x * 16;
    const long long CN3 = (long long)3 * C * N;
    const float* qb = qkv + (long long)b * CN3 + (long long)(h * 48) * N;
    const float* kb = qkv + (long long)b * CN3 + (long long)(C + h * 48) * N;

    const int m  = lane & 15;
    const int kk = (lane < 16) ? 0 : 2;        // 32-bit A 16x4 layout (ISA 7.12.2)
    const float* qrow = qb + (long long)(c0 + m) * N + kk;
    const float* krow = kb + (long long)(d0 + m) * N + kk;

    v8f acc;
    #pragma unroll
    for (int e = 0; e < 8; e++) acc[e] = 0.f;

    for (int n = wave * 4; n < N; n += 32) {
        v2f a, bbv;
        a[0]   = qrow[n]; a[1]   = qrow[n + 1];
        bbv[0] = krow[n]; bbv[1] = krow[n + 1];
        acc = __builtin_amdgcn_wmma_f32_16x16x4_f32(
            false, a, false, bbv, (short)0, acc, false, false);
    }

    __shared__ float red[8][32][8];            // 8 KB
    #pragma unroll
    for (int e = 0; e < 8; e++) red[wave][lane][e] = acc[e];
    __syncthreads();

    int l = t & 31, v = t >> 5;                // one (lane,vgpr) slot per thread
    float s = 0.f;
    #pragma unroll
    for (int w = 0; w < 8; w++) s += red[w][l][v];
    int row = ((l < 16) ? 0 : 8) + v;
    int col = l & 15;
    int c = c0 + row, d = d0 + col;
    float val = s * scale * invq[b * C + h * 48 + c] * invk[b * C + h * 48 + d];
    attn[((long long)bh * 48 + c) * 48 + d] = val;
}

// ---------------------------------------------------------------------------
// Softmax over rows of 48
// ---------------------------------------------------------------------------
__global__ __launch_bounds__(64)
void softmax48(float* __restrict__ attn)
{
    float* p = attn + (long long)blockIdx.x * 48;
    __shared__ float sm[64];
    int t = threadIdx.x;
    float x = (t < 48) ? p[t] : -3.402823466e38f;
    sm[t] = x; __syncthreads();
    for (int off = 32; off > 0; off >>= 1) {
        if (t < off) sm[t] = fmaxf(sm[t], sm[t + off]);
        __syncthreads();
    }
    float mx = sm[0]; __syncthreads();
    float e = (t < 48) ? expf(x - mx) : 0.f;
    sm[t] = e; __syncthreads();
    for (int off = 32; off > 0; off >>= 1) {
        if (t < off) sm[t] += sm[t + off];
        __syncthreads();
    }
    if (t < 48) p[t] = e / sm[0];
}

// ---------------------------------------------------------------------------
// out[c,n] = sum_d attn[c,d] * v[d,n]; attn in LDS, v read once, coalesced.
// Writes into concat channels [0,384).
// ---------------------------------------------------------------------------
__global__ __launch_bounds__(256)
void attn_v(const float* __restrict__ qkv, const float* __restrict__ attn,
            float* __restrict__ out, int C, int N)
{
    __shared__ float a[48][48];
    int t = threadIdx.x;
    int h = blockIdx.y, b = blockIdx.z;
    int n0 = blockIdx.x * 256;
    const float* ap = attn + (long long)(b * 8 + h) * 48 * 48;
    for (int i = t; i < 48 * 48; i += 256) a[i / 48][i % 48] = ap[i];
    __syncthreads();

    const float* vb = qkv + ((long long)b * 3 * C + 2 * C + h * 48) * (long long)N + n0 + t;
    float accv[48];
    #pragma unroll
    for (int c = 0; c < 48; c++) accv[c] = 0.f;
    for (int d = 0; d < 48; d++) {
        float vv = vb[(long long)d * N];
        #pragma unroll
        for (int c = 0; c < 48; c++) accv[c] = fmaf(a[c][d], vv, accv[c]);
    }
    float* ob = out + ((long long)b * 768 + h * 48) * (long long)N + n0 + t;
    for (int c = 0; c < 48; c++) ob[(long long)c * N] = accv[c];
}

// ---------------------------------------------------------------------------
extern "C" void kernel_launch(void* const* d_in, const int* in_sizes, int n_in,
                              void* d_out, int out_size, void* d_ws, size_t ws_size,
                              hipStream_t stream)
{
    const float* x       = (const float*)d_in[0];
    const float* w_qkv   = (const float*)d_in[1];
    const float* w_dw    = (const float*)d_in[2];
    const float* w_proj  = (const float*)d_in[3];
    const float* w_bs_pw = (const float*)d_in[4];
    const float* w_bs_dw = (const float*)d_in[5];
    float* out = (float*)d_out;

    const int B = 4, C = 384, H = 128, W = 128, N = H * W, HEADS = 8;

    char* ws = (char*)d_ws;
    size_t szR1 = (size_t)B * 3 * C * N * sizeof(float);   // 302 MB
    float* R1     = (float*)ws;                 // qkv pre-dw; later concat + bs
    float* R2     = (float*)(ws + szR1);        // qkv post-dw (q,k,v)
    float* attn   = (float*)(ws + 2 * szR1);
    float* invq   = attn + (size_t)B * HEADS * 48 * 48;
    float* invk   = invq + (size_t)B * C;
    float* concat = R1;                         // [b][768][N]
    float* bsbuf  = R1 + (size_t)B * 2 * C * N; // last 100 MB of R1

    // 1. qkv 1x1 conv (bf16 WMMA GEMM, TDM-staged): x -> R1
    gemm_bf16_nn<<<dim3(N / TN, (3 * C) / TM, B), 256, 0, stream>>>(
        w_qkv, x, R1, 3 * C, C, N, (long long)C * N, (long long)3 * C * N);
    // 2. qkv depthwise 3x3: R1 -> R2   (R1 dead after)
    dw3x3<<<dim3(N / 256, 3 * C, B), 256, 0, stream>>>(
        R1, w_dw, R2, 3 * C, H, W, 3 * C, 0, 0);
    // 3. BSConv 1x1 (bf16 WMMA GEMM): x -> bsbuf (aliases tail of R1)
    gemm_bf16_nn<<<dim3(N / TN, C / TM, B), 256, 0, stream>>>(
        w_bs_pw, x, bsbuf, C, C, N, (long long)C * N, (long long)C * N);
    // 4. BSConv depthwise 3x3 + exact GELU -> concat channels [384,768)
    dw3x3<<<dim3(N / 256, C, B), 256, 0, stream>>>(
        bsbuf, w_bs_dw, concat, C, H, W, 2 * C, C, 1);
    // 5. 1/||row|| for q and k
    row_invnorm<<<dim3(B * C, 2), 256, 0, stream>>>(R2, invq, invk, C, N);
    // 6. attn logits (fp32 WMMA 16x16x4), norm scales folded into epilogue
    attn_qk<<<dim3(3, 3, B * HEADS), 256, 0, stream>>>(
        R2, invq, invk, attn, C, N, 1.f / sqrtf(48.f));
    // 7. softmax over d
    softmax48<<<dim3(B * HEADS * 48), 64, 0, stream>>>(attn);
    // 8. attn @ v -> concat channels [0,384)
    attn_v<<<dim3(N / 256, HEADS, B), 256, 0, stream>>>(R2, attn, concat, C, N);
    // 9. proj 1x1 conv (bf16 WMMA GEMM): concat -> out
    gemm_bf16_nn<<<dim3(N / TN, C / TM, B), 256, 0, stream>>>(
        w_proj, concat, out, C, 2 * C, N, (long long)2 * C * N, (long long)C * N);
}
